// PeriodicSAModule_609885356789
// MI455X (gfx1250) — matrix operand order repaired and checked by reading, not compiled, not run on hardware
//
#include <hip/hip_runtime.h>
#include <hip/hip_bf16.h>

// ---------------- problem constants ----------------
#define BATCHN 8
#define NPTS   1024
#define NQ     (BATCHN * NPTS)   // 8192
#define CIN    64
#define HID    128
#define OUTC   128
#define KNBR   64
#define KPADF  96                // padded feat width (67 -> 96), f32 in LDS
#define RADIUS 4.0f
#define INFF   3.402823466e38f

typedef __attribute__((ext_vector_type(16))) _Float16     v16h;
typedef __attribute__((ext_vector_type(8)))  float        v8f;
typedef __attribute__((ext_vector_type(4)))  float        f32x4;
typedef __attribute__((ext_vector_type(4)))  unsigned int u32x4;
typedef __attribute__((ext_vector_type(8)))  unsigned int u32x8;

union FragH {
    v16h  v;
    u32x4 u[2];
};

// workspace layout (bytes)
#define NBR16_BYTES (NQ * KNBR * 2)          // u16 neighbor indices (TDM gather format)
#define MASK_BYTES  (NQ * 2 * 4)             // 2x uint32 validity mask per query
#define W1P_HALVES  (3 * 8 * 512)            // 3 K-chunks x 8 N-tiles x (32 lanes * 16 f16)
#define W2P_HALVES  (4 * 8 * 512)

// ---------------------------------------------------------------------------
// Kernel 0: pack W1/W2 (f32, row-major KxN) into per-lane WMMA B-fragment
// layout.  B 32x16 f16: lanes 0-15 hold col N=lane, K=0..7,16..23 (elems 0..15);
// lanes 16-31 hold col N=lane-16, K=8..15,24..31.
// ---------------------------------------------------------------------------
__global__ void pack_w(const float* __restrict__ W1, const float* __restrict__ W2,
                       _Float16* __restrict__ W1p, _Float16* __restrict__ W2p) {
    int f = blockIdx.x;          // 0..23 -> W1, 24..55 -> W2
    int l = threadIdx.x;         // 0..31
    int kb = (l < 16) ? 0 : 8;
    int nl = l & 15;
    if (f < 24) {
        int kc = f >> 3, nt = f & 7;
        int n = nt * 16 + nl;
        _Float16* dst = W1p + (size_t)f * 512 + l * 16;
#pragma unroll
        for (int e = 0; e < 16; ++e) {
            int k = kc * 32 + kb + (e < 8 ? e : e + 8);
            float v = (k < CIN + 3) ? W1[k * HID + n] : 0.0f;
            dst[e] = (_Float16)v;
        }
    } else {
        int g = f - 24;
        int kc = g >> 3, nt = g & 7;
        int n = nt * 16 + nl;
        _Float16* dst = W2p + (size_t)g * 512 + l * 16;
#pragma unroll
        for (int e = 0; e < 16; ++e) {
            int k = kc * 32 + kb + (e < 8 ? e : e + 8);
            dst[e] = (_Float16)W2[k * OUTC + n];
        }
    }
}

// ---------------------------------------------------------------------------
// Kernel 1: periodic distances + top-64 selection.  One wave32 per query,
// 4 queries per 128-thread block.  Emits u16 neighbor indices (ready for the
// TDM gather descriptor) and a 64-bit radius-validity mask.
// ---------------------------------------------------------------------------
__global__ void topk_kernel(const float* __restrict__ frac,
                            const float* __restrict__ tv,
                            const float* __restrict__ scale,
                            unsigned short* __restrict__ nbr16,
                            unsigned int* __restrict__ maskb) {
    __shared__ float smem[4][NPTS];
    int wave = threadIdx.x >> 5;
    int lane = threadIdx.x & 31;
    int q = blockIdx.x * 4 + wave;      // 0..8191
    int b = q >> 10;

    float T0 = tv[b * 9 + 0], T1 = tv[b * 9 + 1], T2 = tv[b * 9 + 2];
    float T3 = tv[b * 9 + 3], T4 = tv[b * 9 + 4], T5 = tv[b * 9 + 5];
    float T6 = tv[b * 9 + 6], T7 = tv[b * 9 + 7], T8 = tv[b * 9 + 8];
    float fq0 = frac[q * 3 + 0], fq1 = frac[q * 3 + 1], fq2 = frac[q * 3 + 2];

    const float* fb = frac + (size_t)(b << 10) * 3;
    for (int j = lane; j < NPTS; j += 32) {
        float d0 = fb[j * 3 + 0] - fq0;
        float d1 = fb[j * 3 + 1] - fq1;
        float d2 = fb[j * 3 + 2] - fq2;
        float c0 = d0 * T0 + d1 * T3 + d2 * T6;
        float c1 = d0 * T1 + d1 * T4 + d2 * T7;
        float c2 = d0 * T2 + d1 * T5 + d2 * T8;
        smem[wave][j] = c0 * c0 + c1 * c1 + c2 * c2;
    }
    __syncthreads();

    volatile float* sd = &smem[wave][0];
    float rr = RADIUS / scale[b];
    float r2 = rr * rr;
    unsigned int m0 = 0, m1 = 0;

    for (int k = 0; k < KNBR; ++k) {
        float bv = INFF;
        int   bi = NPTS;
        for (int j = lane; j < NPTS; j += 32) {
            float v = sd[j];
            if (v < bv) { bv = v; bi = j; }
        }
#pragma unroll
        for (int off = 16; off > 0; off >>= 1) {
            float ov = __shfl_down(bv, off);
            int   oi = __shfl_down(bi, off);
            if (ov < bv || (ov == bv && oi < bi)) { bv = ov; bi = oi; }
        }
        if (lane == 0) {
            nbr16[(size_t)q * KNBR + k] = (unsigned short)bi;
            if (bv < r2) {
                if (k < 32) m0 |= (1u << k);
                else        m1 |= (1u << (k - 32));
            }
            sd[bi] = INFF;
        }
    }
    if (lane == 0) {
        maskb[q * 2 + 0] = m0;
        maskb[q * 2 + 1] = m1;
    }
}

// ---------------------------------------------------------------------------
// Kernel 2: per-query neighbor MLP with WMMA + masked max aggregation.
// One wave32 per query, all 64 neighbors in a single pass (4 M-tiles), so
// every W fragment is loaded once per query and reused across 4 WMMAs.
//
// Gather: Tensor Data Mover in GATHER mode.  Each descriptor pulls 16 rows of
// x (row width 64 f32, 16-bit row indices) from global into LDS, with LDS
// padding (pad_interval=64 dw, pad_amount=32 dw) landing rows at the 96-f32
// feat stride.  4 descriptors per query, tracked by TENSORcnt.
// ---------------------------------------------------------------------------
__global__ void __launch_bounds__(32)
mlp_kernel(const float* __restrict__ x, const float* __restrict__ pos,
           const float* __restrict__ b1, const float* __restrict__ b2,
           const _Float16* __restrict__ W1p, const _Float16* __restrict__ W2p,
           const unsigned short* __restrict__ nbr16,
           const unsigned int* __restrict__ maskb,
           float* __restrict__ out) {
    __shared__ __attribute__((aligned(16))) float    featf[KNBR * KPADF];  // 24 KB f32
    __shared__ __attribute__((aligned(16))) _Float16 hbuf[KNBR * HID];     // 16 KB f16

    int q = blockIdx.x;
    int b = q >> 10;
    int lane = threadIdx.x;

    __builtin_prefetch(W1p, 0, 3);
    __builtin_prefetch(W2p, 0, 3);

    unsigned int m0 = maskb[q * 2 + 0];
    unsigned int m1 = maskb[q * 2 + 1];

    float px = pos[q * 3 + 0], py = pos[q * 3 + 1], pz = pos[q * 3 + 2];

    // ---- TDM gather: 4 descriptors x 16 neighbor rows ----
    {
        unsigned long long xb =
            (unsigned long long)(uintptr_t)(x + ((size_t)b << 16));  // crystal base
        unsigned ldsbase = (unsigned)(uintptr_t)featf;
        const u32x4* ip = (const u32x4*)(nbr16 + (size_t)q * KNBR);  // packed u16 idx

        // D# group1 (constant across descriptors):
        //   data_size=4B, pad_enable, pad_interval=64dw, pad_amount=32dw,
        //   tensor_dim0=64, tensor_dim1=1024, tile_dim0=64, tile_dim1=16 idx,
        //   tensor_dim0_stride=64
        u32x8 g1;
        g1[0] = (2u << 16) | (1u << 20) | (5u << 22) | (31u << 25);
        g1[1] = (64u << 16);
        g1[2] = (1024u << 16);
        g1[3] = (64u << 16);
        g1[4] = 16u;
        g1[5] = 64u;
        g1[6] = 0u;
        g1[7] = 0u;

#pragma unroll
        for (int t = 0; t < 4; ++t) {
            u32x4 g0;
            g0[0] = 1u | (1u << 31);                 // count=1, gather_mode, 16-bit idx
            g0[1] = ldsbase + t * (16 * KPADF * 4);  // lds_addr (16 rows x 384B)
            g0[2] = (unsigned)xb;                    // global_addr[31:0]
            g0[3] = ((unsigned)(xb >> 32) & 0x01FFFFFFu) | (2u << 30);  // addr[56:32]|type=2
            u32x4 g2 = ip[t * 2 + 0];                // row indices 0..7  (u16 x8)
            u32x4 g3 = ip[t * 2 + 1];                // row indices 8..15 (u16 x8)
            asm volatile("tensor_load_to_lds %0, %1, %2, %3"
                         :: "s"(g0), "s"(g1), "s"(g2), "s"(g3)
                         : "memory");
        }
    }

    // ---- rel-position + zero-pad columns 64..95 while the DMA is in flight ----
#pragma unroll
    for (int rr2 = 0; rr2 < 2; ++rr2) {
        int row = lane + rr2 * 32;
        int j   = nbr16[(size_t)q * KNBR + row];
        int src = (b << 10) + j;
        float* d2 = featf + row * KPADF + 64;
        d2[0] = pos[src * 3 + 0] - px;
        d2[1] = pos[src * 3 + 1] - py;
        d2[2] = pos[src * 3 + 2] - pz;
#pragma unroll
        for (int c = 3; c < 32; ++c) d2[c] = 0.0f;
    }

    __builtin_amdgcn_s_wait_tensorcnt(0);
    __syncthreads();   // single-wave WG: cheap

    int m  = lane & 15;               // A-row within tile this lane owns
    int kb = (lane < 16) ? 0 : 8;     // A K-base per half-wave
    int hi = (lane >> 4) << 3;        // C/D row offset for this half-wave

    float colmax[8];
#pragma unroll
    for (int t = 0; t < 8; ++t) colmax[t] = -INFF;

    // ---- A fragments for GEMM1: 4 M-tiles x 3 K-chunks, f32 LDS -> f16 ----
    v16h a1[4][3];
#pragma unroll
    for (int mt = 0; mt < 4; ++mt) {
        const float* rowf = featf + (mt * 16 + m) * KPADF;
#pragma unroll
        for (int kc = 0; kc < 3; ++kc) {
            const float* p = rowf + kc * 32 + kb;
            f32x4 A = *(const f32x4*)(p);
            f32x4 B = *(const f32x4*)(p + 4);
            f32x4 C = *(const f32x4*)(p + 16);
            f32x4 D = *(const f32x4*)(p + 20);
            v16h a;
#pragma unroll
            for (int e = 0; e < 4; ++e) {
                a[e]      = (_Float16)A[e];
                a[4 + e]  = (_Float16)B[e];
                a[8 + e]  = (_Float16)C[e];
                a[12 + e] = (_Float16)D[e];
            }
            a1[mt][kc] = a;
        }
    }

    // ---- GEMM1: feat[64x96] @ W1[96x128] -> relu -> h (LDS) ----
    // Each B fragment loaded once, reused by 4 M-tiles.
#pragma unroll
    for (int nt = 0; nt < 8; ++nt) {
        FragH fb[3];
#pragma unroll
        for (int kc = 0; kc < 3; ++kc) {
            const _Float16* bp = W1p + (size_t)(kc * 8 + nt) * 512 + lane * 16;
            fb[kc].u[0] = *(const u32x4*)bp;
            fb[kc].u[1] = *(const u32x4*)(bp + 8);
        }
        float bias = b1[nt * 16 + (lane & 15)];
#pragma unroll
        for (int mt = 0; mt < 4; ++mt) {
            v8f acc = {};
#pragma unroll
            for (int kc = 0; kc < 3; ++kc)
                acc = __builtin_amdgcn_wmma_f32_16x16x32_f16(
                        false, a1[mt][kc], false, fb[kc].v, (short)0, acc, false, false);
#pragma unroll
            for (int rr = 0; rr < 8; ++rr) {
                float v = acc[rr] + bias;
                v = v > 0.0f ? v : 0.0f;
                hbuf[(mt * 16 + rr + hi) * HID + nt * 16 + (lane & 15)] = (_Float16)v;
            }
        }
    }
    __syncthreads();

    // ---- A fragments for GEMM2: 4 M-tiles x 4 K-chunks over HID ----
    v16h a2[4][4];
#pragma unroll
    for (int mt = 0; mt < 4; ++mt) {
        const _Float16* hrow = hbuf + (mt * 16 + m) * HID;
#pragma unroll
        for (int kc = 0; kc < 4; ++kc) {
            FragH fa;
            fa.u[0] = *(const u32x4*)(hrow + kc * 32 + kb);
            fa.u[1] = *(const u32x4*)(hrow + kc * 32 + kb + 16);
            a2[mt][kc] = fa.v;
        }
    }

    // ---- GEMM2: h @ W2 -> mask -> running max ----
#pragma unroll
    for (int nt = 0; nt < 8; ++nt) {
        FragH fb[4];
#pragma unroll
        for (int kc = 0; kc < 4; ++kc) {
            const _Float16* bp = W2p + (size_t)(kc * 8 + nt) * 512 + lane * 16;
            fb[kc].u[0] = *(const u32x4*)bp;
            fb[kc].u[1] = *(const u32x4*)(bp + 8);
        }
        float bias = b2[nt * 16 + (lane & 15)];
#pragma unroll
        for (int mt = 0; mt < 4; ++mt) {
            v8f acc = {};
#pragma unroll
            for (int kc = 0; kc < 4; ++kc)
                acc = __builtin_amdgcn_wmma_f32_16x16x32_f16(
                        false, a2[mt][kc], false, fb[kc].v, (short)0, acc, false, false);
#pragma unroll
            for (int rr = 0; rr < 8; ++rr) {
                int kidx = mt * 16 + rr + hi;                // neighbor slot 0..63
                unsigned int mm = (kidx & 32) ? m1 : m0;
                bool valid = (mm >> (kidx & 31)) & 1u;
                float v = valid ? (acc[rr] + bias) : -INFF;
                colmax[nt] = fmaxf(colmax[nt], v);
            }
        }
    }

    // combine the two half-wave row sets, zero if no valid neighbor, write out
    bool any = (m0 | m1) != 0u;
#pragma unroll
    for (int nt = 0; nt < 8; ++nt) {
        float o = __shfl_xor(colmax[nt], 16);
        colmax[nt] = fmaxf(colmax[nt], o);
        if (!any) colmax[nt] = 0.0f;
    }
    if (lane < 16) {
#pragma unroll
        for (int nt = 0; nt < 8; ++nt)
            out[(size_t)q * OUTC + nt * 16 + lane] = colmax[nt];
    }
}

// ---------------------------------------------------------------------------
// Kernel 3: tuple pass-through outputs (pos, fps_pos, batch, frac_pos,
// trans_vec, scale) concatenated after x_out.
// ---------------------------------------------------------------------------
__global__ void pass_kernel(const float* __restrict__ pos, const float* __restrict__ fps,
                            const int* __restrict__ batch, const float* __restrict__ frac,
                            const float* __restrict__ tv, const float* __restrict__ scale,
                            float* __restrict__ out) {
    int i = blockIdx.x * blockDim.x + threadIdx.x;
    const int n0 = NQ * 3;            // pos
    const int n1 = n0 + NQ * 3;       // fps_pos
    const int n2 = n1 + NQ;           // batch
    const int n3 = n2 + NQ * 3;       // frac_pos
    const int n4 = n3 + 72;           // trans_vec
    const int n5 = n4 + BATCHN;       // scale
    if (i < n0)       out[i] = pos[i];
    else if (i < n1)  out[i] = fps[i - n0];
    else if (i < n2)  out[i] = (float)batch[i - n1];
    else if (i < n3)  out[i] = frac[i - n2];
    else if (i < n4)  out[i] = tv[i - n3];
    else if (i < n5)  out[i] = scale[i - n4];
}

// ---------------------------------------------------------------------------
extern "C" void kernel_launch(void* const* d_in, const int* in_sizes, int n_in,
                              void* d_out, int out_size, void* d_ws, size_t ws_size,
                              hipStream_t stream) {
    const float* x     = (const float*)d_in[0];
    const float* pos   = (const float*)d_in[1];
    const float* fps   = (const float*)d_in[2];
    const int*   batch = (const int*)d_in[3];
    const float* frac  = (const float*)d_in[4];
    const float* tv    = (const float*)d_in[5];
    const float* scale = (const float*)d_in[6];
    const float* W1    = (const float*)d_in[7];
    const float* b1    = (const float*)d_in[8];
    const float* W2    = (const float*)d_in[9];
    const float* b2    = (const float*)d_in[10];
    float* out = (float*)d_out;

    char* ws = (char*)d_ws;
    unsigned short* nbr16 = (unsigned short*)ws;
    unsigned int*   maskb = (unsigned int*)(ws + NBR16_BYTES);
    _Float16*       W1p   = (_Float16*)(ws + NBR16_BYTES + MASK_BYTES);
    _Float16*       W2p   = W1p + W1P_HALVES;

    pack_w<<<56, 32, 0, stream>>>(W1, W2, W1p, W2p);
    topk_kernel<<<NQ / 4, 128, 0, stream>>>(frac, tv, scale, nbr16, maskb);
    mlp_kernel<<<NQ, 32, 0, stream>>>(x, pos, b1, b2, W1p, W2p, nbr16, maskb, out);

    const int pass_elems = NQ * 3 * 3 + NQ + 72 + BATCHN;
    pass_kernel<<<(pass_elems + 255) / 256, 256, 0, stream>>>(
        pos, fps, batch, frac, tv, scale, out + (size_t)NQ * OUTC);
}
